// SplitLayer0_0_10462540333325
// MI455X (gfx1250) — matrix operand order repaired and checked by reading, not compiled, not run on hardware
//
#include <hip/hip_runtime.h>
#include <hip/hip_bf16.h>

typedef __attribute__((ext_vector_type(16))) __bf16 v16bf;
typedef __attribute__((ext_vector_type(4)))  __bf16 v4bf;
typedef __attribute__((ext_vector_type(8)))  float  v8f;

#define MODE_LIFT   0
#define MODE_AXPY   1
#define MODE_BNRELU 2

// ---------------------------------------------------------------------------
// Fused GEMM:  out[rows, NCOLS] = f(A)[rows, K] @ B[K, NCOLS]   (bf16 WMMA)
//   MODE_LIFT  : f(A) row e = concat(node[s[e]] + node[t[e]], edge[e])  (K=256)
//                also writes lift_aggr (f32) for the edge branch.
//   MODE_AXPY  : f(A) = (1+eps)*A0 + A1
//   MODE_BNRELU: f(A) = relu(scale[k]*A0 + shift[k])
// Also accumulates per-output-channel sum / sumsq (pre-BN batch stats).
// ---------------------------------------------------------------------------
template<int K, int NCOLS, int MODE>
__global__ __launch_bounds__(256) void gemm_stats_kernel(
    const float* __restrict__ A0, const float* __restrict__ A1,
    const float* __restrict__ edgeR,
    const int* __restrict__ sidx, const int* __restrict__ tidx,
    const float* __restrict__ epsp,
    const float* __restrict__ bnScale, const float* __restrict__ bnShift,
    const __bf16* __restrict__ Bsw,
    float* __restrict__ out, float* __restrict__ liftOut,
    float* __restrict__ statSum, float* __restrict__ statSumSq,
    int rows)
{
    constexpr int TM     = (K > 128) ? 64 : 128;  // row tile (keeps LDS at 32KB)
    constexpr int F      = K / 32;                // k-steps per WMMA chain
    constexpr int NT     = NCOLS / 16;            // 16-wide col tiles
    constexpr int GROUPS = TM / 16;               // 16-row groups per block
    constexpr int WPG    = 8 / GROUPS;            // waves sharing a row group
    constexpr int NT_PER = NT / WPG;
    constexpr int QUADS  = TM * K / 4;

    __shared__ alignas(32) __bf16 tileA[TM * K];  // fragment-swizzled A
    __shared__ float lsum[NCOLS];
    __shared__ float lsumsq[NCOLS];

    const int tid = threadIdx.x;
    for (int i = tid; i < NCOLS; i += 256) { lsum[i] = 0.f; lsumsq[i] = 0.f; }

    const int rowBase = blockIdx.x * TM;
    float alphav = 0.f;
    if (MODE == MODE_AXPY) alphav = 1.0f + epsp[0];

    // ---- cooperative load: f32 quads -> elementwise op -> bf16 swizzled LDS
    for (int q = tid; q < QUADS; q += 256) {
        const int r   = q / (K / 4);
        const int kk  = (q % (K / 4)) * 4;
        const int row = rowBase + r;
        float4 a = make_float4(0.f, 0.f, 0.f, 0.f);
        if (row < rows) {
            if (MODE == MODE_LIFT) {
                if (kk < 128) {
                    const float4 sv = *(const float4*)&A0[(size_t)sidx[row] * 128 + kk];
                    const float4 tv = *(const float4*)&A0[(size_t)tidx[row] * 128 + kk];
                    a = make_float4(sv.x + tv.x, sv.y + tv.y, sv.z + tv.z, sv.w + tv.w);
                    *(float4*)&liftOut[(size_t)row * 128 + kk] = a;
                } else {
                    a = *(const float4*)&edgeR[(size_t)row * 128 + (kk - 128)];
                }
            } else if (MODE == MODE_AXPY) {
                const size_t o = (size_t)row * K + kk;
                const float4 x = *(const float4*)&A0[o];
                const float4 y = *(const float4*)&A1[o];
                a = make_float4(fmaf(alphav, x.x, y.x), fmaf(alphav, x.y, y.y),
                                fmaf(alphav, x.z, y.z), fmaf(alphav, x.w, y.w));
            } else {
                const size_t o  = (size_t)row * K + kk;
                const float4 x  = *(const float4*)&A0[o];
                const float4 sc = *(const float4*)&bnScale[kk];
                const float4 sh = *(const float4*)&bnShift[kk];
                a = make_float4(fmaxf(fmaf(sc.x, x.x, sh.x), 0.f),
                                fmaxf(fmaf(sc.y, x.y, sh.y), 0.f),
                                fmaxf(fmaf(sc.z, x.z, sh.z), 0.f),
                                fmaxf(fmaf(sc.w, x.w, sh.w), 0.f));
            }
        }
        // ISA 7.12.2 16-bit A 16x32 layout: lane = h*16+m holds
        // K = (v/8)*16 + h*8 + (v%8), v in 0..15.  kk%4==0 keeps v..v+3 contiguous.
        const int g   = r >> 4, m = r & 15;
        const int f   = kk >> 5, kin = kk & 31;
        const int h   = (kin & 15) >> 3;
        const int v   = ((kin >> 4) << 3) | (kin & 7);
        const int off = (((g * F + f) << 5) + (h * 16 + m)) * 16 + v;
        *(v4bf*)&tileA[off] = v4bf{(__bf16)a.x, (__bf16)a.y, (__bf16)a.z, (__bf16)a.w};
    }
    __syncthreads();

    const int wave = tid >> 5, lane = tid & 31;
    const int g    = wave / WPG, sub = wave % WPG;

    // hoist all A fragments for this wave's 16 rows (full K) into VGPRs
    v16bf afr[F];
#pragma unroll
    for (int f = 0; f < F; ++f)
        afr[f] = *(const v16bf*)&tileA[(((g * F + f) << 5) + lane) * 16];

    const int rowTile = rowBase + g * 16;
    const int colLane = lane & 15;
    const int rOff    = (lane >> 4) << 3;    // D layout: lanes16-31 hold M=8..15
    const bool full   = (rowTile + 16) <= rows;

#pragma unroll
    for (int t = 0; t < NT_PER; ++t) {
        const int nt = sub * NT_PER + t;
        v8f acc = {0.f, 0.f, 0.f, 0.f, 0.f, 0.f, 0.f, 0.f};
#pragma unroll
        for (int f = 0; f < F; ++f) {
            const v16bf bfr =
                *(const v16bf*)&Bsw[(((size_t)f * NT + nt) * 32 + lane) * 16];
            acc = __builtin_amdgcn_wmma_f32_16x16x32_bf16(
                false, afr[f], false, bfr, (short)0, acc, false, false);
        }
        const int col = nt * 16 + colLane;
        float* op = out + (size_t)(rowTile + rOff) * NCOLS + col;
        float s = 0.f, sq = 0.f;
        if (full) {                       // branch-free dominant path
#pragma unroll
            for (int r = 0; r < 8; ++r) {
                const float d = acc[r];
                op[(size_t)r * NCOLS] = d;
                s += d; sq += d * d;
            }
        } else {
#pragma unroll
            for (int r = 0; r < 8; ++r) {
                if (rowTile + rOff + r < rows) {
                    const float d = acc[r];
                    op[(size_t)r * NCOLS] = d;
                    s += d; sq += d * d;
                }
            }
        }
        atomicAdd(&lsum[col], s);         // ds_add_f32
        atomicAdd(&lsumsq[col], sq);
    }
    __syncthreads();
    for (int i = tid; i < NCOLS; i += 256) {
        atomicAdd(&statSum[i],   lsum[i]);
        atomicAdd(&statSumSq[i], lsumsq[i]);
    }
}

// ---------------------------------------------------------------------------
// Pre-swizzle a f32 weight matrix [K, NCOLS] into the bf16 B-fragment layout:
// element (tile=kt*NT+nt, lane=l, v) = W[kt*32 + (v/8)*16 + (l/16)*8 + v%8][nt*16 + l%16]
// so each lane's 16 B values are one contiguous 32B load.
// ---------------------------------------------------------------------------
__global__ void swizzle_weight_kernel(const float* __restrict__ W,
                                      __bf16* __restrict__ Bsw,
                                      int K, int NCOLS) {
    const int idx = blockIdx.x * blockDim.x + threadIdx.x;
    if (idx >= K * NCOLS) return;
    const int NT = NCOLS / 16;
    const int v    = idx & 15;
    const int l    = (idx >> 4) & 31;
    const int tile = idx >> 9;
    const int nt   = tile % NT;
    const int kt   = tile / NT;
    const int n    = nt * 16 + (l & 15);
    const int k    = kt * 32 + (v >> 3) * 16 + (l >> 4) * 8 + (v & 7);
    Bsw[idx] = (__bf16)W[(size_t)k * NCOLS + n];
}

__global__ void bn_finalize_kernel(const float* __restrict__ sum,
                                   const float* __restrict__ sumsq,
                                   const float* __restrict__ gamma,
                                   const float* __restrict__ beta,
                                   float invCount, int C,
                                   float* __restrict__ scale,
                                   float* __restrict__ shift) {
    const int i = blockIdx.x * blockDim.x + threadIdx.x;
    if (i >= C) return;
    const float mu  = sum[i] * invCount;
    const float var = sumsq[i] * invCount - mu * mu;
    const float sc  = gamma[i] * rsqrtf(var + 1e-5f);
    scale[i] = sc;
    shift[i] = beta[i] - mu * sc;
}

// h = bn_relu(X1) per element-quad; scatter-add into both endpoints.
__global__ void bn_relu_scatter_kernel(const float* __restrict__ X,
                                       const float* __restrict__ scale,
                                       const float* __restrict__ shift,
                                       const int* __restrict__ sidx,
                                       const int* __restrict__ tidx,
                                       float* __restrict__ lvl, int E) {
    const size_t totalQ = (size_t)E * 32;            // quads of 4 channels
    const size_t stride = (size_t)gridDim.x * blockDim.x;
    for (size_t q = blockIdx.x * (size_t)blockDim.x + threadIdx.x;
         q < totalQ; q += stride) {
        const int e = (int)(q >> 5);
        const int c = (int)(q & 31) * 4;
        const float4 x  = *(const float4*)&X[(size_t)e * 128 + c];
        const float4 sc = *(const float4*)&scale[c];
        const float4 sh = *(const float4*)&shift[c];
        const float h0 = fmaxf(fmaf(sc.x, x.x, sh.x), 0.f);
        const float h1 = fmaxf(fmaf(sc.y, x.y, sh.y), 0.f);
        const float h2 = fmaxf(fmaf(sc.z, x.z, sh.z), 0.f);
        const float h3 = fmaxf(fmaf(sc.w, x.w, sh.w), 0.f);
        float* ls = lvl + (size_t)sidx[e] * 128 + c;
        float* lt = lvl + (size_t)tidx[e] * 128 + c;
        atomicAdd(ls + 0, h0); atomicAdd(ls + 1, h1);
        atomicAdd(ls + 2, h2); atomicAdd(ls + 3, h3);
        atomicAdd(lt + 0, h0); atomicAdd(lt + 1, h1);
        atomicAdd(lt + 2, h2); atomicAdd(lt + 3, h3);
    }
}

__global__ void bn_relu_out_kernel(const float* __restrict__ X,
                                   const float* __restrict__ scale,
                                   const float* __restrict__ shift,
                                   float* __restrict__ out, size_t totalQ) {
    const size_t stride = (size_t)gridDim.x * blockDim.x;
    for (size_t q = blockIdx.x * (size_t)blockDim.x + threadIdx.x;
         q < totalQ; q += stride) {
        const int c = (int)(q & 31) * 4;
        const float4 x  = *(const float4*)&X[q * 4];
        const float4 sc = *(const float4*)&scale[c];
        const float4 sh = *(const float4*)&shift[c];
        *(float4*)&out[q * 4] = make_float4(
            fmaxf(fmaf(sc.x, x.x, sh.x), 0.f), fmaxf(fmaf(sc.y, x.y, sh.y), 0.f),
            fmaxf(fmaf(sc.z, x.z, sh.z), 0.f), fmaxf(fmaf(sc.w, x.w, sh.w), 0.f));
    }
}

__global__ void zero_kernel(float* __restrict__ p, size_t nQ) {
    const size_t stride = (size_t)gridDim.x * blockDim.x;
    for (size_t i = blockIdx.x * (size_t)blockDim.x + threadIdx.x; i < nQ; i += stride)
        *(float4*)&p[i * 4] = make_float4(0.f, 0.f, 0.f, 0.f);
}

// ---------------------------------------------------------------------------
extern "C" void kernel_launch(void* const* d_in, const int* in_sizes, int n_in,
                              void* d_out, int out_size, void* d_ws, size_t ws_size,
                              hipStream_t stream) {
    (void)n_in; (void)out_size; (void)ws_size;
    const float* node_rep  = (const float*)d_in[0];
    const float* edge_rep  = (const float*)d_in[1];
    const int*   endpoints = (const int*)d_in[2];
    const float* W_lvl1 = (const float*)d_in[3];
    const float* g_lvl1 = (const float*)d_in[4];
    const float* b_lvl1 = (const float*)d_in[5];
    const float* W_n1   = (const float*)d_in[6];
    const float* g_n1   = (const float*)d_in[7];
    const float* b_n1   = (const float*)d_in[8];
    const float* W_n2   = (const float*)d_in[9];
    const float* g_n2   = (const float*)d_in[10];
    const float* b_n2   = (const float*)d_in[11];
    const float* W_e1   = (const float*)d_in[12];
    const float* g_e1   = (const float*)d_in[13];
    const float* b_e1   = (const float*)d_in[14];
    const float* W_e2   = (const float*)d_in[15];
    const float* g_e2   = (const float*)d_in[16];
    const float* b_e2   = (const float*)d_in[17];
    const float* eps1   = (const float*)d_in[18];
    const float* eps2   = (const float*)d_in[19];

    const int C = 128;
    const int N = in_sizes[0] / C;
    const int E = in_sizes[1] / C;
    const int* sIdx = endpoints;
    const int* tIdx = endpoints + E;

    // workspace layout (f32 unless noted)
    float*  bufH    = (float*)d_ws;                       // [E,128]: X1, later X_n2/X_e2
    float*  bufLift = bufH    + (size_t)E * 128;          // [E,128]: lift_aggr
    float*  bufBig  = bufLift + (size_t)E * 128;          // [E,256]: hidden (node then edge)
    float*  bufLvl  = bufBig  + (size_t)E * 256;          // [N,128]: lvl_aggr
    float*  stats   = bufLvl  + (size_t)N * 128;          // 5 stages x 1024 f32
    __bf16* wsw     = (__bf16*)(stats + 5 * 1024);        // 5 x 32768 bf16 swizzled weights

    // 1) swizzle weights to bf16 B-fragment order
    struct WD { const float* W; int K, Nc; } wd[5] = {
        {W_lvl1, 256, 128}, {W_n1, 128, 256}, {W_n2, 256, 128},
        {W_e1, 128, 256},   {W_e2, 256, 128}};
    for (int i = 0; i < 5; ++i) {
        const int elems = wd[i].K * wd[i].Nc;   // 32768 each
        swizzle_weight_kernel<<<(elems + 255) / 256, 256, 0, stream>>>(
            wd[i].W, wsw + (size_t)i * 32768, wd[i].K, wd[i].Nc);
    }
    // 2) zero accumulators
    zero_kernel<<<2048, 256, 0, stream>>>(bufLvl, (size_t)N * 32);
    zero_kernel<<<8, 256, 0, stream>>>(stats, (5 * 1024) / 4);

    float* s0 = stats + 0 * 1024;  // [sum | sumsq | scale | shift] each 256 slots
    float* s1 = stats + 1 * 1024;
    float* s2 = stats + 2 * 1024;
    float* s3 = stats + 3 * 1024;
    float* s4 = stats + 4 * 1024;

    // 3) stage 0: gather + concat + GEMM (E x 256 x 128) -> X1, lift_aggr
    gemm_stats_kernel<256, 128, MODE_LIFT><<<(E + 63) / 64, 256, 0, stream>>>(
        node_rep, nullptr, edge_rep, sIdx, tIdx, nullptr, nullptr, nullptr,
        wsw + 0 * 32768, bufH, bufLift, s0, s0 + 256, E);
    bn_finalize_kernel<<<1, 128, 0, stream>>>(s0, s0 + 256, g_lvl1, b_lvl1,
                                              1.0f / E, 128, s0 + 512, s0 + 768);
    // 4) h = bn_relu(X1); scatter-add into lvl_aggr at both endpoints
    bn_relu_scatter_kernel<<<8192, 256, 0, stream>>>(
        bufH, s0 + 512, s0 + 768, sIdx, tIdx, bufLvl, E);

    // 5) node branch
    gemm_stats_kernel<128, 256, MODE_AXPY><<<(N + 127) / 128, 256, 0, stream>>>(
        node_rep, bufLvl, nullptr, nullptr, nullptr, eps1, nullptr, nullptr,
        wsw + 1 * 32768, bufBig, nullptr, s1, s1 + 256, N);
    bn_finalize_kernel<<<1, 256, 0, stream>>>(s1, s1 + 256, g_n1, b_n1,
                                              1.0f / N, 256, s1 + 512, s1 + 768);
    gemm_stats_kernel<256, 128, MODE_BNRELU><<<(N + 63) / 64, 256, 0, stream>>>(
        bufBig, nullptr, nullptr, nullptr, nullptr, nullptr, s1 + 512, s1 + 768,
        wsw + 2 * 32768, bufH, nullptr, s2, s2 + 256, N);
    bn_finalize_kernel<<<1, 128, 0, stream>>>(s2, s2 + 256, g_n2, b_n2,
                                              1.0f / N, 128, s2 + 512, s2 + 768);
    bn_relu_out_kernel<<<4096, 256, 0, stream>>>(
        bufH, s2 + 512, s2 + 768, (float*)d_out, (size_t)N * 32);

    // 6) edge branch
    gemm_stats_kernel<128, 256, MODE_AXPY><<<(E + 127) / 128, 256, 0, stream>>>(
        edge_rep, bufLift, nullptr, nullptr, nullptr, eps2, nullptr, nullptr,
        wsw + 3 * 32768, bufBig, nullptr, s3, s3 + 256, E);
    bn_finalize_kernel<<<1, 256, 0, stream>>>(s3, s3 + 256, g_e1, b_e1,
                                              1.0f / E, 256, s3 + 512, s3 + 768);
    gemm_stats_kernel<256, 128, MODE_BNRELU><<<(E + 63) / 64, 256, 0, stream>>>(
        bufBig, nullptr, nullptr, nullptr, nullptr, nullptr, s3 + 512, s3 + 768,
        wsw + 4 * 32768, bufH, nullptr, s4, s4 + 256, E);
    bn_finalize_kernel<<<1, 128, 0, stream>>>(s4, s4 + 256, g_e2, b_e2,
                                              1.0f / E, 128, s4 + 512, s4 + 768);
    bn_relu_out_kernel<<<8192, 256, 0, stream>>>(
        bufH, s4 + 512, s4 + 768, (float*)d_out + (size_t)N * 128, (size_t)E * 32);
}
